// VideoDiT_37512244363708
// MI455X (gfx1250) — 8-bit matrix operand order repaired and checked by reading, compiled, not run on hardware
//
#include <hip/hip_runtime.h>
#include <stdint.h>

#define DD    1024
#define NHh   16
#define HDh   64
#define Tn    16
#define HWn   256
#define Lseq  4096
#define NCc   128
#define RANKr 16
#define FREQf 256

typedef __attribute__((ext_vector_type(8))) int v8i;

#if __has_builtin(__builtin_amdgcn_global_load_async_to_lds_b128) && \
    __has_builtin(__builtin_amdgcn_s_wait_asynccnt)
#define USE_ASYNC_LDS 1
typedef int v4i_gcc __attribute__((vector_size(16)));
typedef __attribute__((address_space(1))) v4i_gcc* gp128;
typedef __attribute__((address_space(3))) v4i_gcc* lp128;
#else
#define USE_ASYNC_LDS 0
#endif

static constexpr size_t KB = 1024;
static constexpr size_t MB = 1024 * 1024;
static constexpr size_t O_QW   = 0;
static constexpr size_t O_SC   = O_QW   + 24 * MB;
static constexpr size_t O_INVY = O_SC   + 256;
static constexpr size_t O_INVC = O_INVY + 16 * KB;
static constexpr size_t O_INVB = O_INVC + 512;
static constexpr size_t O_D1   = O_INVB + 16 * KB;
static constexpr size_t O_POSH = O_D1   + 256 * KB;
static constexpr size_t O_POS  = O_POSH + 64 * KB;
static constexpr size_t O_Q8A  = O_POS  + 64 * KB;
static constexpr size_t O_Q8B  = O_Q8A  + 4 * MB;
static constexpr size_t O_Q8C  = O_Q8B  + 4 * MB;
static constexpr size_t O_XB   = O_Q8C  + 128 * KB;
static constexpr size_t O_B1   = O_XB   + 16 * MB;
static constexpr size_t O_B2   = O_B1   + 16 * MB;
static constexpr size_t O_B3   = O_B2   + 16 * MB;
static constexpr size_t O_B4   = O_B3   + 16 * MB;
static constexpr size_t O_Y    = O_B4   + 16 * MB;
static constexpr size_t O_KV   = O_Y    + 16 * MB;
static constexpr size_t O_SCR  = O_KV   + 32 * MB;

__global__ void abssum_k(const float* __restrict__ w, float* __restrict__ sum, int n) {
    float v = 0.f;
    for (int i = blockIdx.x * blockDim.x + threadIdx.x; i < n; i += gridDim.x * blockDim.x)
        v += fabsf(w[i]);
    for (int off = 16; off > 0; off >>= 1) v += __shfl_down(v, off, 32);
    if ((threadIdx.x & 31) == 0) atomicAdd(sum, v);
}

__global__ void wquant_k(const float* __restrict__ w, const float* __restrict__ sum,
                         int8_t* __restrict__ t, float* __restrict__ scale, int n) {
    const float mean = fmaxf(*sum / (float)n, 1e-5f);
    const float s = 1.f / mean;
    int i = blockIdx.x * blockDim.x + threadIdx.x;
    if (i == 0) *scale = mean;
    for (; i < n; i += gridDim.x * blockDim.x) {
        float v = rintf(w[i] * s);
        t[i] = (int8_t)fminf(fmaxf(v, -1.f), 1.f);
    }
}

__global__ __launch_bounds__(256) void rmsnorm_quant_k(const float* __restrict__ x,
                                                       int8_t* __restrict__ q,
                                                       float* __restrict__ inva) {
    __shared__ float red[256];
    const int row = blockIdx.x;
    const int tid = threadIdx.x;
    const float* xr = x + (size_t)row * DD;
    float ss = 0.f;
    for (int j = tid; j < DD; j += 256) { float v = xr[j]; ss += v * v; }
    red[tid] = ss; __syncthreads();
    for (int s = 128; s > 0; s >>= 1) { if (tid < s) red[tid] += red[tid + s]; __syncthreads(); }
    const float rms = rsqrtf(red[0] / (float)DD + 1e-6f);
    __syncthreads();
    float am = 0.f;
    for (int j = tid; j < DD; j += 256) am = fmaxf(am, fabsf(xr[j] * rms));
    red[tid] = am; __syncthreads();
    for (int s = 128; s > 0; s >>= 1) { if (tid < s) red[tid] = fmaxf(red[tid], red[tid + s]); __syncthreads(); }
    const float s127 = 127.f / fmaxf(red[0], 1e-5f);
    if (tid == 0) inva[row] = 1.f / s127;
    for (int j = tid; j < DD; j += 256) {
        float v = rintf(xr[j] * rms * s127);
        q[(size_t)row * DD + j] = (int8_t)fminf(fmaxf(v, -128.f), 127.f);
    }
}

#define BM 64
#define BNr 256
#define BK 64
#define LDT 80

__global__ __launch_bounds__(256) void gemm_i8_wmma(const int8_t* __restrict__ A,
                                                    const float* __restrict__ ainv,
                                                    const int8_t* __restrict__ W,
                                                    const float* __restrict__ wsc,
                                                    float* __restrict__ C,
                                                    int M, int N, int K) {
    __shared__ int8_t lA[2][BM * LDT];
    __shared__ int8_t lB[2][BNr * LDT];

    const int tid  = threadIdx.x;
    const int m0   = blockIdx.y * BM;
    const int n0   = blockIdx.x * BNr;
    const int lane = tid & 31;
    const int wave = tid >> 5;
    const int wm   = (wave & 1) * 32;
    const int wn   = (wave >> 1) * 64;
    const int lr   = lane & 15;
    const int hi   = lane >> 4;

    const int arow = tid >> 2;
    const int acol = (tid & 3) * 16;

    auto stage = [&](int buf, int kt) {
        const int8_t* ga = A + (size_t)(m0 + arow) * K + kt + acol;
        int8_t* la = &lA[buf][arow * LDT + acol];
#if USE_ASYNC_LDS
        __builtin_amdgcn_global_load_async_to_lds_b128((gp128)ga, (lp128)la, 0, 0);
#else
        *(int4*)la = *(const int4*)ga;
#endif
        const int8_t* gb = W + (size_t)(n0 + tid) * K + kt;
        int8_t* lb = &lB[buf][tid * LDT];
#pragma unroll
        for (int cb = 0; cb < 4; ++cb) {
#if USE_ASYNC_LDS
            __builtin_amdgcn_global_load_async_to_lds_b128((gp128)(gb + 16 * cb),
                                                           (lp128)(lb + 16 * cb), 0, 0);
#else
            *(int4*)(lb + 16 * cb) = *(const int4*)(gb + 16 * cb);
#endif
        }
    };

    v8i acc[2][4];
#pragma unroll
    for (int i = 0; i < 2; ++i)
#pragma unroll
        for (int j = 0; j < 4; ++j)
#pragma unroll
            for (int r = 0; r < 8; ++r) acc[i][j][r] = 0;

    stage(0, 0);
    int cur = 0;
    for (int kt = 0; kt < K; kt += BK) {
#if USE_ASYNC_LDS
        __builtin_amdgcn_s_wait_asynccnt(0);
#endif
        __syncthreads();
        if (kt + BK < K) {
            stage(cur ^ 1, kt + BK);
#if !USE_ASYNC_LDS
            __builtin_prefetch(A + (size_t)(m0 + arow) * K + kt + BK + acol, 0, 1);
#endif
        }

        v8i af[2], bf[4];
#pragma unroll
        for (int f = 0; f < 2; ++f) {
            const int8_t* ab = &lA[cur][(wm + f * 16 + lr) * LDT + hi * 16];
#pragma unroll
            for (int c = 0; c < 4; ++c) {
                int2 dv = *(const int2*)(ab + 8 * (c & 1) + 32 * (c >> 1));
                af[f][2 * c]     = dv.x;
                af[f][2 * c + 1] = dv.y;
            }
        }
#pragma unroll
        for (int j = 0; j < 4; ++j) {
            const int8_t* bb = &lB[cur][(wn + j * 16 + lr) * LDT + hi * 16];
            int4 p0 = *(const int4*)(bb);
            int4 p1 = *(const int4*)(bb + 32);
            bf[j][0] = p0.x; bf[j][1] = p0.y; bf[j][2] = p0.z; bf[j][3] = p0.w;
            bf[j][4] = p1.x; bf[j][5] = p1.y; bf[j][6] = p1.z; bf[j][7] = p1.w;
        }
#pragma unroll
        for (int i = 0; i < 2; ++i)
#pragma unroll
            for (int j = 0; j < 4; ++j)
                acc[i][j] = __builtin_amdgcn_wmma_i32_16x16x64_iu8(
                    true, af[i], true, bf[j], acc[i][j], false, false);
        cur ^= 1;
    }

    const float wsv = *wsc;
#pragma unroll
    for (int i = 0; i < 2; ++i)
#pragma unroll
        for (int j = 0; j < 4; ++j)
#pragma unroll
            for (int r = 0; r < 8; ++r) {
                const int m = m0 + wm + i * 16 + r + hi * 8;
                const int n = n0 + wn + j * 16 + lr;
                C[(size_t)m * N + n] = (float)acc[i][j][r] * ainv[m] * wsv;
            }
}

__global__ void gate_k(float* __restrict__ ib, float* __restrict__ fb, int n) {
    int i = blockIdx.x * blockDim.x + threadIdx.x;
    if (i >= n) return;
    const float fl = fb[i], il = ib[i];
    const float f  = 1.f / (1.f + __expf(-fl));
    const float si = il / (1.f + __expf(-il));
    fb[i] = f;
    ib[i] = si * (1.f - f);
}

__global__ void scan_k(const float* __restrict__ f, const float* __restrict__ iv,
                       float* __restrict__ h, int nseq, int tlen) {
    int id = blockIdx.x * blockDim.x + threadIdx.x;
    if (id >= nseq * DD) return;
    const int s = id / DD, d = id % DD;
    float hv = 0.f;
    const size_t base = (size_t)s * tlen * DD + d;
    for (int t = 0; t < tlen; ++t) {
        const size_t idx = base + (size_t)t * DD;
        hv = f[idx] * hv + iv[idx];
        h[idx] = hv;
    }
}

__global__ __launch_bounds__(64) void headgate_k(float* __restrict__ h,
                                                 const float* __restrict__ g,
                                                 const float* __restrict__ gn) {
    __shared__ float red[64];
    const int row  = blockIdx.x / NHh;
    const int head = blockIdx.x % NHh;
    const int t    = threadIdx.x;
    const size_t idx = (size_t)row * DD + head * HDh + t;
    const float hv = h[idx];
    red[t] = hv * hv; __syncthreads();
    for (int s = 32; s > 0; s >>= 1) { if (t < s) red[t] += red[t + s]; __syncthreads(); }
    const float rms = rsqrtf(red[0] / (float)HDh + 1e-6f);
    const float gv  = g[idx];
    const float sg  = gv / (1.f + __expf(-gv));
    h[idx] = hv * rms * gn[t] * sg;
}

__global__ void copy_k(float* __restrict__ o, const float* __restrict__ a, int n) {
    int i = blockIdx.x * blockDim.x + threadIdx.x;
    if (i < n) o[i] = a[i];
}
__global__ void add_k(float* __restrict__ o, const float* __restrict__ a, int n) {
    int i = blockIdx.x * blockDim.x + threadIdx.x;
    if (i < n) o[i] += a[i];
}
__global__ void add2_k(float* __restrict__ o, const float* __restrict__ a,
                       const float* __restrict__ b, int n) {
    int i = blockIdx.x * blockDim.x + threadIdx.x;
    if (i < n) o[i] = a[i] + b[i];
}

__global__ void t_gather_k(const float* __restrict__ x, const float* __restrict__ pos,
                           float* __restrict__ xt, int n) {
    int id = blockIdx.x * blockDim.x + threadIdx.x;
    if (id >= n) return;
    const int d = id % DD, row = id / DD;
    const int t = row % Tn, hw = row / Tn;
    xt[id] = x[((size_t)(t * HWn + hw)) * DD + d] + pos[t * DD + d];
}
__global__ void t_scatter_add_k(float* __restrict__ y, const float* __restrict__ ot, int n) {
    int id = blockIdx.x * blockDim.x + threadIdx.x;
    if (id >= n) return;
    const int d = id % DD, row = id / DD;
    const int t = row % Tn, hw = row / Tn;
    y[((size_t)(t * HWn + hw)) * DD + d] += ot[id];
}
__global__ void col_gather_k(const float* __restrict__ x, float* __restrict__ xc, int n) {
    int id = blockIdx.x * blockDim.x + threadIdx.x;
    if (id >= n) return;
    const int d = id % DD, r = id / DD;
    const int h = r % 16, w = (r / 16) % 16, t = r / 256;
    xc[id] = x[((size_t)(t * 256 + h * 16 + w)) * DD + d];
}
__global__ void col_scatter_add_k(float* __restrict__ y, const float* __restrict__ oc, int n) {
    int id = blockIdx.x * blockDim.x + threadIdx.x;
    if (id >= n) return;
    const int d = id % DD, r = id / DD;
    const int h = r % 16, w = (r / 16) % 16, t = r / 256;
    y[((size_t)(t * 256 + h * 16 + w)) * DD + d] += oc[id];
}

__global__ void pos1_k(const float* __restrict__ fw1, const float* __restrict__ fb1,
                       float* __restrict__ hid) {
    int id = blockIdx.x * blockDim.x + threadIdx.x;
    if (id >= Tn * DD) return;
    const int j = id % DD, t = id / DD;
    const float ti = (float)t / (float)Tn;
    float acc = fb1[j];
    for (int f = 0; f < FREQf / 2; ++f) {
        const float fr  = __expf(-logf(10000.f) * (float)f / (float)(FREQf / 2));
        const float arg = ti * fr * 1000.f;
        acc += cosf(arg) * fw1[(size_t)j * FREQf + f];
        acc += sinf(arg) * fw1[(size_t)j * FREQf + FREQf / 2 + f];
    }
    hid[id] = acc;
}
__global__ void pos2_k(const float* __restrict__ hid, const float* __restrict__ fw2,
                       const float* __restrict__ fb2, float* __restrict__ pos) {
    int id = blockIdx.x * blockDim.x + threadIdx.x;
    if (id >= Tn * DD) return;
    const int j = id % DD, t = id / DD;
    float acc = fb2[j];
    for (int i2 = 0; i2 < DD; ++i2) {
        const float hv = hid[t * DD + i2];
        acc += (hv / (1.f + __expf(-hv))) * fw2[(size_t)j * DD + i2];
    }
    pos[id] = acc;
}

__global__ void lora1_k(const float* __restrict__ qin, const float* __restrict__ la,
                        float* __restrict__ d1, int M) {
    int id = blockIdx.x * blockDim.x + threadIdx.x;
    if (id >= M * RANKr) return;
    const int r = id % RANKr, m = id / RANKr;
    float acc = 0.f;
    for (int k = 0; k < DD; ++k) acc += qin[(size_t)m * DD + k] * la[(size_t)r * DD + k];
    d1[id] = acc;
}
__global__ void lora2_addmag_k(float* __restrict__ qp, const float* __restrict__ d1,
                               const float* __restrict__ lb, const float* __restrict__ mag,
                               int M) {
    int id = blockIdx.x * blockDim.x + threadIdx.x;
    if (id >= M * DD) return;
    const int j = id % DD, m = id / DD;
    float acc = 0.f;
    for (int r = 0; r < RANKr; ++r) acc += d1[(size_t)m * RANKr + r] * lb[(size_t)j * RANKr + r];
    qp[id] += mag[j / HDh] * acc;
}
__global__ void scores_k(const float* __restrict__ qf, const float* __restrict__ kv,
                         float* __restrict__ sc, int M, int Nc) {
    int id = blockIdx.x * blockDim.x + threadIdx.x;
    const int total = NHh * M * Nc;
    if (id >= total) return;
    const int k = id % Nc, q = (id / Nc) % M, h = id / (Nc * M);
    const float* qv = qf + (size_t)q * DD + h * HDh;
    const float* kr = kv + (size_t)k * (2 * DD) + h * HDh;
    float acc = 0.f;
#pragma unroll 8
    for (int d = 0; d < HDh; ++d) acc += qv[d] * kr[d];
    sc[id] = acc * 0.125f;
}
__global__ __launch_bounds__(256) void softmax_k(float* __restrict__ sc, int Nc) {
    __shared__ float red[256];
    float* row = sc + (size_t)blockIdx.x * Nc;
    const int tid = threadIdx.x;
    float mx = -1e30f;
    for (int k = tid; k < Nc; k += 256) mx = fmaxf(mx, row[k]);
    red[tid] = mx; __syncthreads();
    for (int s = 128; s > 0; s >>= 1) { if (tid < s) red[tid] = fmaxf(red[tid], red[tid + s]); __syncthreads(); }
    mx = red[0]; __syncthreads();
    float sm = 0.f;
    for (int k = tid; k < Nc; k += 256) { const float e = __expf(row[k] - mx); row[k] = e; sm += e; }
    red[tid] = sm; __syncthreads();
    for (int s = 128; s > 0; s >>= 1) { if (tid < s) red[tid] += red[tid + s]; __syncthreads(); }
    const float inv = 1.f / red[0];
    for (int k = tid; k < Nc; k += 256) row[k] *= inv;
}
__global__ void attnout_k(const float* __restrict__ sc, const float* __restrict__ kv,
                          float* __restrict__ out, int M, int Nc) {
    int id = blockIdx.x * blockDim.x + threadIdx.x;
    if (id >= M * DD) return;
    const int j = id % DD, q = id / DD;
    const int h = j / HDh;
    const float* pr = sc + ((size_t)h * M + q) * Nc;
    float acc = 0.f;
    for (int k = 0; k < Nc; ++k) acc += pr[k] * kv[(size_t)k * (2 * DD) + DD + j];
    out[id] = acc;
}

static inline int cdiv(int a, int b) { return (a + b - 1) / b; }

extern "C" void kernel_launch(void* const* d_in, const int* in_sizes, int n_in,
                              void* d_out, int out_size, void* d_ws, size_t ws_size,
                              hipStream_t stream) {
    (void)in_sizes; (void)n_in; (void)out_size; (void)ws_size;
    char* ws = (char*)d_ws;

    const float* x    = (const float*)d_in[0];
    const float* c    = (const float*)d_in[1];
    const float* hw[4] = { (const float*)d_in[2], (const float*)d_in[3],
                           (const float*)d_in[4], (const float*)d_in[5] };
    const float* hgn  = (const float*)d_in[6];
    const float* fw1  = (const float*)d_in[7];
    const float* fb1  = (const float*)d_in[8];
    const float* fw2  = (const float*)d_in[9];
    const float* fb2  = (const float*)d_in[10];
    const float* wq_x  = (const float*)d_in[11];
    const float* wkv_c = (const float*)d_in[12];
    const float* mag_x = (const float*)d_in[13];
    const float* la_x  = (const float*)d_in[14];
    const float* lb_x  = (const float*)d_in[15];
    const float* wp_x  = (const float*)d_in[16];
    const float* wq_c  = (const float*)d_in[17];
    const float* wkv_x = (const float*)d_in[18];
    const float* mag_c = (const float*)d_in[19];
    const float* la_c  = (const float*)d_in[20];
    const float* lb_c  = (const float*)d_in[21];
    const float* wp_c  = (const float*)d_in[22];

    float*  SC   = (float*)(ws + O_SC);
    float*  INVY = (float*)(ws + O_INVY);
    float*  INVC = (float*)(ws + O_INVC);
    float*  INVB = (float*)(ws + O_INVB);
    float*  D1   = (float*)(ws + O_D1);
    float*  POSH = (float*)(ws + O_POSH);
    float*  POS  = (float*)(ws + O_POS);
    int8_t* Q8A  = (int8_t*)(ws + O_Q8A);
    int8_t* Q8B  = (int8_t*)(ws + O_Q8B);
    int8_t* Q8C  = (int8_t*)(ws + O_Q8C);
    float*  XB   = (float*)(ws + O_XB);
    float*  B1   = (float*)(ws + O_B1);
    float*  B2   = (float*)(ws + O_B2);
    float*  B3   = (float*)(ws + O_B3);
    float*  B4   = (float*)(ws + O_B4);
    float*  Y    = (float*)(ws + O_Y);
    float*  KV   = (float*)(ws + O_KV);
    float*  SCR  = (float*)(ws + O_SCR);
    float*  out  = (float*)d_out;

    const int ND = Lseq * DD;
    const int EB = cdiv(ND, 256);

    (void)hipMemsetAsync(ws + O_SC, 0, 256, stream);
    struct WT { const float* src; size_t qoff; int n; };
    WT wt[22];
    int wid = 0;
    for (int m = 0; m < 4; ++m)
        for (int p = 0; p < 4; ++p) {
            wt[wid] = { hw[m] + (size_t)p * DD * DD, O_QW + (size_t)(m * 4 + p) * MB, DD * DD };
            ++wid;
        }
    wt[16] = { wq_x,  O_QW + 16 * MB, DD * DD };
    wt[17] = { wkv_c, O_QW + 17 * MB, 2 * DD * DD };
    wt[18] = { wp_x,  O_QW + 19 * MB, DD * DD };
    wt[19] = { wq_c,  O_QW + 20 * MB, DD * DD };
    wt[20] = { wkv_x, O_QW + 21 * MB, 2 * DD * DD };
    wt[21] = { wp_c,  O_QW + 23 * MB, DD * DD };
    for (int i = 0; i < 22; ++i) {
        abssum_k<<<512, 256, 0, stream>>>(wt[i].src, SC + 2 * i, wt[i].n);
        wquant_k<<<1024, 256, 0, stream>>>(wt[i].src, SC + 2 * i,
                                           (int8_t*)(ws + wt[i].qoff), SC + 2 * i + 1, wt[i].n);
    }

    auto gemm = [&](const int8_t* Aq, const float* ainv, int widx, float* Cp, int M, int N) {
        dim3 g(N / BNr, M / BM);
        gemm_i8_wmma<<<g, 256, 0, stream>>>(Aq, ainv, (const int8_t*)(ws + wt[widx].qoff),
                                            SC + 2 * widx + 1, Cp, M, N, DD);
    };

    pos1_k<<<cdiv(Tn * DD, 256), 256, 0, stream>>>(fw1, fb1, POSH);
    pos2_k<<<cdiv(Tn * DD, 256), 256, 0, stream>>>(POSH, fw2, fb2, POS);

    copy_k<<<EB, 256, 0, stream>>>(Y, x, ND);

    auto run_hgrn = [&](const float* xin, int p, int nseq, int tlen) {
        rmsnorm_quant_k<<<Lseq, 256, 0, stream>>>(xin, Q8A, INVY);
        gemm(Q8A, INVY, 0 * 4 + p, B1, Lseq, DD);
        gemm(Q8A, INVY, 1 * 4 + p, B2, Lseq, DD);
        gemm(Q8A, INVY, 2 * 4 + p, B3, Lseq, DD);
        gate_k<<<EB, 256, 0, stream>>>(B1, B2, ND);
        scan_k<<<cdiv(nseq * DD, 256), 256, 0, stream>>>(B2, B1, B4, nseq, tlen);
        headgate_k<<<Lseq * NHh, 64, 0, stream>>>(B4, B3, hgn + (size_t)p * HDh);
        rmsnorm_quant_k<<<Lseq, 256, 0, stream>>>(B4, Q8A, INVY);
        gemm(Q8A, INVY, 3 * 4 + p, B1, Lseq, DD);
    };

    run_hgrn(x, 0, Tn, HWn);
    add_k<<<EB, 256, 0, stream>>>(Y, B1, ND);
    t_gather_k<<<EB, 256, 0, stream>>>(x, POS, XB, ND);
    run_hgrn(XB, 1, HWn, Tn);
    t_scatter_add_k<<<EB, 256, 0, stream>>>(Y, B1, ND);
    col_gather_k<<<EB, 256, 0, stream>>>(x, XB, ND);
    run_hgrn(XB, 2, 256, 16);
    col_scatter_add_k<<<EB, 256, 0, stream>>>(Y, B1, ND);
    run_hgrn(x, 2, 256, 16);
    add_k<<<EB, 256, 0, stream>>>(Y, B1, ND);
    run_hgrn(x, 3, 1, Lseq);
    add_k<<<EB, 256, 0, stream>>>(Y, B1, ND);

    rmsnorm_quant_k<<<Lseq, 256, 0, stream>>>(Y, Q8A, INVY);
    rmsnorm_quant_k<<<NCc, 256, 0, stream>>>(c, Q8C, INVC);

    gemm(Q8A, INVY, 16, B2, Lseq, DD);
    gemm(Q8C, INVC, 17, KV, NCc, 2 * DD);
    lora1_k<<<cdiv(Lseq * RANKr, 256), 256, 0, stream>>>(Y, la_x, D1, Lseq);
    lora2_addmag_k<<<EB, 256, 0, stream>>>(B2, D1, lb_x, mag_x, Lseq);
    scores_k<<<cdiv(NHh * Lseq * NCc, 256), 256, 0, stream>>>(B2, KV, SCR, Lseq, NCc);
    softmax_k<<<NHh * Lseq, 256, 0, stream>>>(SCR, NCc);
    attnout_k<<<EB, 256, 0, stream>>>(SCR, KV, B3, Lseq, NCc);
    rmsnorm_quant_k<<<Lseq, 256, 0, stream>>>(B3, Q8B, INVB);
    gemm(Q8B, INVB, 18, B4, Lseq, DD);
    add2_k<<<EB, 256, 0, stream>>>(out, Y, B4, ND);

    gemm(Q8C, INVC, 19, B2, NCc, DD);
    gemm(Q8A, INVY, 20, KV, Lseq, 2 * DD);
    lora1_k<<<cdiv(NCc * RANKr, 256), 256, 0, stream>>>(c, la_c, D1, NCc);
    lora2_addmag_k<<<cdiv(NCc * DD, 256), 256, 0, stream>>>(B2, D1, lb_c, mag_c, NCc);
    scores_k<<<cdiv(NHh * NCc * Lseq, 256), 256, 0, stream>>>(B2, KV, SCR, NCc, Lseq);
    softmax_k<<<NHh * NCc, 256, 0, stream>>>(SCR, Lseq);
    attnout_k<<<cdiv(NCc * DD, 256), 256, 0, stream>>>(SCR, KV, B3, NCc, Lseq);
    rmsnorm_quant_k<<<NCc, 256, 0, stream>>>(B3, Q8B, INVB);
    gemm(Q8B, INVB, 21, B4, NCc, DD);
    add2_k<<<cdiv(NCc * DD, 256), 256, 0, stream>>>(out + (size_t)Lseq * DD, c, B4, NCc * DD);
}
